// PNPPNDS_69956427317650
// MI455X (gfx1250) — matrix operand order repaired and checked by reading, not compiled, not run on hardware
//
#include <hip/hip_runtime.h>
#include <hip/hip_bf16.h>
#include <stdint.h>

// ---------------------------------------------------------------------------
// FlowNet3D-ish pipeline for gfx1250 (MI455X).
//  - FPS: persistent single-workgroup kernel, wave32 shuffle argmax
//  - Ball query: LDS-tiled brute force, first-K-in-radius semantics
//  - All GEMMs: v_wmma_f32_16x16x32_f16, one 16x16 D tile per wave
//  - Train-mode BN: stats kernel (atomics) + apply kernel
// ---------------------------------------------------------------------------

#define NPTS   22000
#define KSA    16
#define KFE    24
#define R2SA   0.49f
#define R2FE   1.0f
#define EPSF   1e-5f

typedef _Float16 v8h  __attribute__((ext_vector_type(8)));
typedef _Float16 v16h __attribute__((ext_vector_type(16)));
typedef float    v8f  __attribute__((ext_vector_type(8)));

// ---------------------------------------------------------------------------
// Farthest point sampling. pos is channel-major [3, Np]. One workgroup,
// 1024 threads (32 wave32s). dist lives in global (L2-resident, 88KB).
// ---------------------------------------------------------------------------
__global__ __launch_bounds__(1024)
void fps_kernel(const float* __restrict__ pos, int Np, float* __restrict__ dist,
                int* __restrict__ outIdx, int npoint) {
  const int tid  = threadIdx.x;
  const int lane = tid & 31;
  const int wave = tid >> 5;
  __shared__ float rv[32];
  __shared__ int   ri[32];
  __shared__ int   sFar;
  for (int j = tid; j < Np; j += 1024) dist[j] = 1e10f;
  if (tid == 0) sFar = 0;
  __syncthreads();
  int far = 0;
  for (int it = 0; it < npoint; ++it) {
    if (tid == 0) outIdx[it] = far;
    const float cx = pos[far], cy = pos[Np + far], cz = pos[2 * Np + far];
    float bv = -1.0f; int bi = 0;
    for (int j = tid; j < Np; j += 1024) {
      float dx = pos[j] - cx, dy = pos[Np + j] - cy, dz = pos[2 * Np + j] - cz;
      float d  = dx * dx + dy * dy + dz * dz;
      float nd = fminf(dist[j], d);
      dist[j]  = nd;
      if (nd > bv) { bv = nd; bi = j; }
    }
    // wave32 argmax reduce (ties -> smaller index, matching jnp.argmax)
    for (int s = 16; s > 0; s >>= 1) {
      float ov = __shfl_xor(bv, s, 32);
      int   oi = __shfl_xor(bi, s, 32);
      if (ov > bv || (ov == bv && oi < bi)) { bv = ov; bi = oi; }
    }
    if (lane == 0) { rv[wave] = bv; ri[wave] = bi; }
    __syncthreads();
    if (wave == 0) {
      float v2 = rv[lane]; int i2 = ri[lane];
      for (int s = 16; s > 0; s >>= 1) {
        float ov = __shfl_xor(v2, s, 32);
        int   oi = __shfl_xor(i2, s, 32);
        if (ov > v2 || (ov == v2 && oi < i2)) { v2 = ov; i2 = oi; }
      }
      if (lane == 0) sFar = i2;
    }
    __syncthreads();
    far = sFar;
  }
}

__global__ void gather_pos(const float* __restrict__ pos, const int* __restrict__ idx,
                           float* __restrict__ npos, int Np, int S) {
  int t = blockIdx.x * 256 + threadIdx.x;
  if (t >= 3 * S) return;
  int c = t / S, s = t - c * S;
  npos[c * S + s] = pos[c * Np + idx[s]];
}

// ---------------------------------------------------------------------------
// Ball query: queries q [3,S] against points pos [3,Np]; first K (ascending
// index) points within radius; pad with first hit; Np-1 when no neighbor.
// ---------------------------------------------------------------------------
#define BQ_TILE 1024
__global__ __launch_bounds__(256)
void ball_query_kernel(const float* __restrict__ pos, int Np,
                       const float* __restrict__ q, int S,
                       float r2, int K, int* __restrict__ out) {
  __shared__ float tx[BQ_TILE], ty[BQ_TILE], tz[BQ_TILE];
  int s = blockIdx.x * 256 + threadIdx.x;
  bool active = s < S;
  float qx = 0.f, qy = 0.f, qz = 0.f;
  if (active) { qx = q[s]; qy = q[S + s]; qz = q[2 * S + s]; }
  int cnt = 0, first = Np - 1;
  for (int base = 0; base < Np; base += BQ_TILE) {
    int tn = (Np - base < BQ_TILE) ? (Np - base) : BQ_TILE;
    for (int j = threadIdx.x; j < tn; j += 256) {
      tx[j] = pos[base + j]; ty[j] = pos[Np + base + j]; tz[j] = pos[2 * Np + base + j];
      if (base + BQ_TILE + j < Np) __builtin_prefetch(&pos[base + BQ_TILE + j], 0, 0);
    }
    __syncthreads();
    if (active && cnt < K) {
      for (int j = 0; j < tn; ++j) {
        float dx = tx[j] - qx, dy = ty[j] - qy, dz = tz[j] - qz;
        float d = dx * dx + dy * dy + dz * dz;
        if (d <= r2) {
          if (cnt == 0) first = base + j;
          out[(size_t)s * K + cnt] = base + j;
          if (++cnt == K) break;
        }
      }
    }
    __syncthreads();
  }
  if (active) for (int k = cnt; k < K; ++k) out[(size_t)s * K + k] = first;
}

// ---------------------------------------------------------------------------
// Build f16 input matrices (row-major [M, CK], zero-padded channels)
// ---------------------------------------------------------------------------
__global__ void build_A_sa(const float* __restrict__ pos, const float* __restrict__ npos,
                           const float* __restrict__ feats, const int* __restrict__ idx,
                           _Float16* __restrict__ A, int Np, int S, int K, int C, int CK) {
  int t = blockIdx.x * 256 + threadIdx.x;
  if (t >= S * K) return;
  int s = t / K;
  int j = idx[t];
  _Float16* row = A + (size_t)t * CK;
  row[0] = (_Float16)(pos[j]          - npos[s]);
  row[1] = (_Float16)(pos[Np + j]     - npos[S + s]);
  row[2] = (_Float16)(pos[2 * Np + j] - npos[2 * S + s]);
  if (feats) { for (int c = 0; c < C; ++c) row[3 + c] = (_Float16)feats[(size_t)c * Np + j]; }
  else       { for (int c = 0; c < C; ++c) row[3 + c] = (_Float16)0.0f; }
  for (int p = 3 + C; p < CK; ++p) row[p] = (_Float16)0.0f;
}

__global__ void build_A_fe(const float* __restrict__ p1, const float* __restrict__ p2,
                           const float* __restrict__ f1, const float* __restrict__ f2,
                           const int* __restrict__ idx, _Float16* __restrict__ A,
                           int Nn, int K, int C, int CK) {
  int t = blockIdx.x * 256 + threadIdx.x;
  if (t >= Nn * K) return;
  int n = t / K;
  int j = idx[t];
  _Float16* row = A + (size_t)t * CK;
  row[0] = (_Float16)(p2[j]          - p1[n]);
  row[1] = (_Float16)(p2[Nn + j]     - p1[Nn + n]);
  row[2] = (_Float16)(p2[2 * Nn + j] - p1[2 * Nn + n]);
  for (int c = 0; c < C; ++c) {
    row[3 + c]     = (_Float16)f2[(size_t)c * Nn + j];
    row[3 + C + c] = (_Float16)f1[(size_t)c * Nn + n];
  }
  for (int p = 3 + 2 * C; p < CK; ++p) row[p] = (_Float16)0.0f;
}

__global__ void build_head_A(const float* __restrict__ fa, const float* __restrict__ fb,
                             const float* __restrict__ fc, _Float16* __restrict__ A, int Nn) {
  size_t t = (size_t)blockIdx.x * 256 + threadIdx.x;
  if (t >= (size_t)Nn * 224) return;
  int c = (int)(t % 224);
  int n = (int)(t / 224);
  float v;
  if      (c < 32) v = fa[(size_t)c * Nn + n];
  else if (c < 96) v = fb[(size_t)(c - 32) * Nn + n];
  else             v = fc[(size_t)(c - 96) * Nn + n];
  A[t] = (_Float16)v;
}

// ---------------------------------------------------------------------------
// Pack W [O, Cin] f32 -> B fragments for v_wmma_f32_16x16x32_f16.
// Tile (kt, to): 32 lanes x 16 halves; lane l holds B[k,n] for
// n = to*16 + (l&15), k = kt*32 + (l>>4)*8 + (h&7) + (h>>3)*16  (zero for k >= Cin).
// ---------------------------------------------------------------------------
__global__ void repack_w(const float* __restrict__ W, _Float16* __restrict__ Bp,
                         int Cin, int CK, int O) {
  int t = blockIdx.x * 256 + threadIdx.x;
  int tilesO = O >> 4;
  int total = (CK >> 5) * tilesO * 512;
  if (t >= total) return;
  int h    = t & 15;
  int lane = (t >> 4) & 31;
  int tile = t >> 9;
  int to = tile % tilesO;
  int kt = tile / tilesO;
  int n = to * 16 + (lane & 15);
  int k = kt * 32 + ((lane >> 4) << 3) + (h & 7) + ((h >> 3) << 4);
  Bp[t] = (k < Cin) ? (_Float16)W[(size_t)n * Cin + k] : (_Float16)0.0f;
}

// ---------------------------------------------------------------------------
// Y[M,O] = A[M,CK] x W^T (+bias, optional leaky 0.2). One 16x16 tile / wave.
// A fragment per the ISA 16-bit A 16x32 layout, B from packed tiles,
// C/D per the 32-bit 16x16 layout (M = r + 8*(lane>>4), N = lane&15).
// ---------------------------------------------------------------------------
__global__ __launch_bounds__(256)
void wmma_gemm(const _Float16* __restrict__ A, const _Float16* __restrict__ Bp,
               const float* __restrict__ bias, float* __restrict__ Y,
               int M, int CK, int O, int leaky) {
  const int lane = threadIdx.x & 31;
  const int gw = blockIdx.x * 8 + (threadIdx.x >> 5);
  const int tilesO = O >> 4;
  const int totalTiles = (M >> 4) * tilesO;
  if (gw >= totalTiles) return;
  const int to = gw % tilesO;
  const int tm = gw / tilesO;
  const int m  = (tm << 4) + (lane & 15);
  const int kb = (lane >> 4) << 3;            // half offset within 32-wide K chunk
  const _Float16* __restrict__ Arow = A + (size_t)m * CK;
  v8f c = {};
  const int nkt = CK >> 5;
  for (int kt = 0; kt < nkt; ++kt) {
    const v8h a0 = *(const v8h*)(Arow + (kt << 5) + kb);
    const v8h a1 = *(const v8h*)(Arow + (kt << 5) + kb + 16);
    const _Float16* bt = Bp + (((size_t)kt * tilesO + to) << 9) + (lane << 4);
    const v8h b0 = *(const v8h*)bt;
    const v8h b1 = *(const v8h*)(bt + 8);
    v16h av, bv;
#pragma unroll
    for (int i = 0; i < 8; ++i) { av[i] = a0[i]; av[i + 8] = a1[i];
                                  bv[i] = b0[i]; bv[i + 8] = b1[i]; }
    c = __builtin_amdgcn_wmma_f32_16x16x32_f16(false, av, false, bv, (short)0, c,
                                               false, false);
  }
  const int o  = (to << 4) + (lane & 15);
  const float bb = bias ? bias[o] : 0.0f;
  const int mr = (tm << 4) + ((lane >> 4) << 3);
#pragma unroll
  for (int r = 0; r < 8; ++r) {
    float v = c[r] + bb;
    if (leaky) v = (v > 0.0f) ? v : 0.2f * v;
    Y[(size_t)(mr + r) * O + o] = v;
  }
}

// ---------------------------------------------------------------------------
// Train-mode BN machinery
// ---------------------------------------------------------------------------
__global__ void fill_f32(float* __restrict__ p, int n, float v) {
  int t = blockIdx.x * 256 + threadIdx.x;
  if (t < n) p[t] = v;
}

__global__ void bn_stats(const float* __restrict__ Y, float* __restrict__ sums,
                         int M, int O) {
  int gt = blockIdx.x * 256 + threadIdx.x;   // grid = O*4 blocks -> O*1024 threads
  int o = gt % O;
  int r = gt / O;
  float s = 0.f, q = 0.f;
  for (int m = r; m < M; m += 1024) {
    float v = Y[(size_t)m * O + o];
    s += v; q += v * v;
  }
  atomicAdd(&sums[o], s);
  atomicAdd(&sums[O + o], q);
}

__global__ void bn_apply_f16(const float* __restrict__ Y, const float* __restrict__ sums,
                             const float* __restrict__ g, const float* __restrict__ be,
                             _Float16* __restrict__ A2, int M, int O, int CKo, int relu) {
  size_t t = (size_t)blockIdx.x * 256 + threadIdx.x;
  if (t >= (size_t)M * CKo) return;
  int o = (int)(t % CKo);
  size_t m = t / CKo;
  float v = 0.f;
  if (o < O) {
    float mean = sums[o] / (float)M;
    float var  = sums[O + o] / (float)M - mean * mean;
    float sc = g[o] * rsqrtf(var + EPSF);
    v = (Y[m * O + o] - mean) * sc + be[o];
    if (relu) v = fmaxf(v, 0.f);
  }
  A2[t] = (_Float16)v;
}

__global__ void bn_relu_maxk(const float* __restrict__ Y, const float* __restrict__ sums,
                             const float* __restrict__ g, const float* __restrict__ be,
                             float* __restrict__ feats, int S, int K, int O, int M) {
  int t = blockIdx.x * 256 + threadIdx.x;
  if (t >= S * O) return;
  int o = t % O;
  int s = t / O;
  float mean = sums[o] / (float)M;
  float var  = sums[O + o] / (float)M - mean * mean;
  float sc = g[o] * rsqrtf(var + EPSF);
  float sh = be[o] - mean * sc;
  float best = -1e30f;
  for (int k = 0; k < K; ++k) {
    float v = Y[((size_t)s * K + k) * O + o] * sc + sh;
    v = fmaxf(v, 0.0f);
    best = fmaxf(best, v);
  }
  feats[(size_t)o * S + s] = best;
}

// Final 128->3 conv + residual add of x[:,3]
__global__ void head_final(const _Float16* __restrict__ A, const float* __restrict__ W,
                           const float* __restrict__ b, const float* __restrict__ x3,
                           float* __restrict__ out, int Nn, int C) {
  int n = blockIdx.x * 256 + threadIdx.x;
  if (n >= Nn) return;
  float a0 = b[0], a1 = b[1], a2 = b[2];
  const _Float16* row = A + (size_t)n * C;
  for (int c = 0; c < C; ++c) {
    float a = (float)row[c];
    a0 += a * W[c]; a1 += a * W[C + c]; a2 += a * W[2 * C + c];
  }
  out[n]          = x3[n]          + a0;
  out[Nn + n]     = x3[Nn + n]     + a1;
  out[2 * Nn + n] = x3[2 * Nn + n] + a2;
}

// ---------------------------------------------------------------------------
// Host-side orchestration
// ---------------------------------------------------------------------------
struct Ws {
  float* dist; int* fps; int* bq; float* sums;
  _Float16* Bp; _Float16* A; _Float16* A2; float* Y;
};

static inline unsigned cdiv(size_t a, size_t b) { return (unsigned)((a + b - 1) / b); }

static void run_gemm(hipStream_t st, const Ws& w, const _Float16* A, int M, int CK,
                     const float* W, const float* bias, int Cin, int O, int leaky,
                     float* Y) {
  int tilesO = O >> 4;
  int nhalves = (CK >> 5) * tilesO * 512;
  repack_w<<<dim3(cdiv(nhalves, 256)), dim3(256), 0, st>>>(W, w.Bp, Cin, CK, O);
  int totalTiles = (M >> 4) * tilesO;
  wmma_gemm<<<dim3(cdiv(totalTiles, 8)), dim3(256), 0, st>>>(A, w.Bp, bias, Y, M, CK, O, leaky);
}

static void run_stats(hipStream_t st, const Ws& w, const float* Y, int M, int O) {
  fill_f32<<<dim3(cdiv(2 * O, 256)), dim3(256), 0, st>>>(w.sums, 2 * O, 0.f);
  bn_stats<<<dim3((unsigned)(O * 4)), dim3(256), 0, st>>>(Y, w.sums, M, O);
}

static void run_sa(hipStream_t st, const Ws& w, const float* pos, const float* feats, int C,
                   const float* W1, const float* b1, const float* g1, const float* be1,
                   int CK1, int O1,
                   const float* W2, const float* b2, const float* g2, const float* be2, int O2,
                   float* nposOut, float* featsOut) {
  const int Np = NPTS, S = NPTS;
  fps_kernel<<<dim3(1), dim3(1024), 0, st>>>(pos, Np, w.dist, w.fps, S);
  gather_pos<<<dim3(cdiv(3 * S, 256)), dim3(256), 0, st>>>(pos, w.fps, nposOut, Np, S);
  ball_query_kernel<<<dim3(cdiv(S, 256)), dim3(256), 0, st>>>(pos, Np, nposOut, S, R2SA, KSA, w.bq);
  const int M = S * KSA;
  build_A_sa<<<dim3(cdiv(M, 256)), dim3(256), 0, st>>>(pos, nposOut, feats, w.bq, w.A,
                                                       Np, S, KSA, C, CK1);
  run_gemm(st, w, w.A, M, CK1, W1, b1, 3 + C, O1, 0, w.Y);
  run_stats(st, w, w.Y, M, O1);
  if (W2) {
    bn_apply_f16<<<dim3(cdiv((size_t)M * O1, 256)), dim3(256), 0, st>>>(
        w.Y, w.sums, g1, be1, w.A2, M, O1, O1, 1);
    run_gemm(st, w, w.A2, M, O1, W2, b2, O1, O2, 0, w.Y);
    run_stats(st, w, w.Y, M, O2);
    bn_relu_maxk<<<dim3(cdiv((size_t)S * O2, 256)), dim3(256), 0, st>>>(
        w.Y, w.sums, g2, be2, featsOut, S, KSA, O2, M);
  } else {
    bn_relu_maxk<<<dim3(cdiv((size_t)S * O1, 256)), dim3(256), 0, st>>>(
        w.Y, w.sums, g1, be1, featsOut, S, KSA, O1, M);
  }
}

static void run_fe(hipStream_t st, const Ws& w, const float* p1, const float* p2,
                   const float* f1, const float* f2, int C,
                   const float* W, const float* b, const float* g, const float* be,
                   int CK, int O, float* featsOut) {
  const int Nn = NPTS;
  ball_query_kernel<<<dim3(cdiv(Nn, 256)), dim3(256), 0, st>>>(p2, Nn, p1, Nn, R2FE, KFE, w.bq);
  const int M = Nn * KFE;
  build_A_fe<<<dim3(cdiv(M, 256)), dim3(256), 0, st>>>(p1, p2, f1, f2, w.bq, w.A, Nn, KFE, C, CK);
  run_gemm(st, w, w.A, M, CK, W, b, 3 + 2 * C, O, 0, w.Y);
  run_stats(st, w, w.Y, M, O);
  bn_relu_maxk<<<dim3(cdiv((size_t)Nn * O, 256)), dim3(256), 0, st>>>(
      w.Y, w.sums, g, be, featsOut, Nn, KFE, O, M);
}

extern "C" void kernel_launch(void* const* d_in, const int* in_sizes, int n_in,
                              void* d_out, int out_size, void* d_ws, size_t ws_size,
                              hipStream_t stream) {
  (void)in_sizes; (void)out_size; (void)ws_size;
  const float* P[43];
  for (int i = 0; i < 43 && i < n_in; ++i) P[i] = (const float*)d_in[i];
  const float* x = P[0];                                  // [4, 3, N]

  // ---- bump allocator over d_ws ----
  char* base = (char*)d_ws;
  size_t off = 0;
  auto alloc = [&](size_t nbytes) -> void* {
    off = (off + 255) & ~(size_t)255;
    void* p = base + off;
    off += nbytes;
    return p;
  };
  const size_t MFE = (size_t)NPTS * KFE;                  // 528000
  Ws w;
  w.dist = (float*)alloc((size_t)NPTS * 4);
  w.fps  = (int*)  alloc((size_t)NPTS * 4);
  w.bq   = (int*)  alloc(MFE * 4);
  w.sums = (float*)alloc(1024 * 4);
  w.Bp   = (_Float16*)alloc((size_t)262144 * 2);
  w.A    = (_Float16*)alloc(MFE * 160 * 2);               // 169 MB
  w.A2   = (_Float16*)alloc((size_t)NPTS * KSA * 64 * 2); // 45 MB
  w.Y    = (float*)alloc(MFE * 64 * 4);                   // 135 MB

  float *np1[4], *np2[2], *np3, *sa1f[4], *fe1f[2], *sa2f[2], *fe2f, *sa3f;
  for (int t = 0; t < 4; ++t) np1[t] = (float*)alloc((size_t)3 * NPTS * 4);
  for (int t = 0; t < 2; ++t) np2[t] = (float*)alloc((size_t)3 * NPTS * 4);
  np3 = (float*)alloc((size_t)3 * NPTS * 4);
  for (int t = 0; t < 4; ++t) sa1f[t] = (float*)alloc((size_t)32 * NPTS * 4);
  for (int t = 0; t < 2; ++t) fe1f[t] = (float*)alloc((size_t)32 * NPTS * 4);
  for (int t = 0; t < 2; ++t) sa2f[t] = (float*)alloc((size_t)64 * NPTS * 4);
  fe2f = (float*)alloc((size_t)64 * NPTS * 4);
  sa3f = (float*)alloc((size_t)128 * NPTS * 4);

  // ---- SA1 on the 4 frames (feats = zeros5) ----
  for (int t = 0; t < 4; ++t)
    run_sa(stream, w, x + (size_t)t * 3 * NPTS, nullptr, 5,
           P[1], P[2], P[3], P[4], 32, 32,
           P[5], P[6], P[7], P[8], 32,
           np1[t], sa1f[t]);

  // ---- Flow embedding 1 (pairs (1,0) and (3,2)): Cin = 3+2*32 = 67 -> 96 ----
  run_fe(stream, w, np1[1], np1[0], sa1f[1], sa1f[0], 32,
         P[9], P[10], P[11], P[12], 96, 32, fe1f[0]);
  run_fe(stream, w, np1[3], np1[2], sa1f[3], sa1f[2], 32,
         P[9], P[10], P[11], P[12], 96, 32, fe1f[1]);

  // ---- SA2 x2 (Cin = 35 -> 64) ----
  run_sa(stream, w, np1[1], fe1f[0], 32,
         P[13], P[14], P[15], P[16], 64, 64,
         P[17], P[18], P[19], P[20], 64, np2[0], sa2f[0]);
  run_sa(stream, w, np1[3], fe1f[1], 32,
         P[13], P[14], P[15], P[16], 64, 64,
         P[17], P[18], P[19], P[20], 64, np2[1], sa2f[1]);

  // ---- Flow embedding 2 (Cin = 3+2*64 = 131 -> 160) ----
  run_fe(stream, w, np2[1], np2[0], sa2f[1], sa2f[0], 64,
         P[21], P[22], P[23], P[24], 160, 64, fe2f);

  // ---- SA3 (Cin = 67 -> 96, single layer, O=128) ----
  run_sa(stream, w, np2[1], fe2f, 64,
         P[25], P[26], P[27], P[28], 96, 128,
         nullptr, nullptr, nullptr, nullptr, 0, np3, sa3f);

  // ---- Head: aggr [N,224] -> 512 -> 256 -> 128 -> 3 (+ residual) ----
  build_head_A<<<dim3(cdiv((size_t)NPTS * 224, 256)), dim3(256), 0, stream>>>(
      sa1f[3], sa2f[1], sa3f, w.A, NPTS);

  run_gemm(stream, w, w.A, NPTS, 224, P[29], P[30], 224, 512, 1, w.Y);
  run_stats(stream, w, w.Y, NPTS, 512);
  bn_apply_f16<<<dim3(cdiv((size_t)NPTS * 512, 256)), dim3(256), 0, stream>>>(
      w.Y, w.sums, P[37], P[38], w.A2, NPTS, 512, 512, 0);

  run_gemm(stream, w, w.A2, NPTS, 512, P[31], P[32], 512, 256, 1, w.Y);
  run_stats(stream, w, w.Y, NPTS, 256);
  bn_apply_f16<<<dim3(cdiv((size_t)NPTS * 256, 256)), dim3(256), 0, stream>>>(
      w.Y, w.sums, P[39], P[40], w.A, NPTS, 256, 256, 0);

  run_gemm(stream, w, w.A, NPTS, 256, P[33], P[34], 256, 128, 1, w.Y);
  run_stats(stream, w, w.Y, NPTS, 128);
  bn_apply_f16<<<dim3(cdiv((size_t)NPTS * 128, 256)), dim3(256), 0, stream>>>(
      w.Y, w.sums, P[41], P[42], w.A2, NPTS, 128, 128, 0);

  head_final<<<dim3(cdiv(NPTS, 256)), dim3(256), 0, stream>>>(
      w.A2, P[35], P[36], x + (size_t)9 * NPTS, (float*)d_out, NPTS, 128);
}
